// PatchcoreModel_78700980732320
// MI455X (gfx1250) — matrix operand order repaired and checked by reading, not compiled, not run on hardware
//
#include <hip/hip_runtime.h>
#include <hip/hip_bf16.h>

typedef __attribute__((ext_vector_type(16))) __bf16        v16bf;
typedef __attribute__((ext_vector_type(8)))  float         v8f;
typedef __attribute__((ext_vector_type(4)))  unsigned int  u32x4;
typedef __attribute__((ext_vector_type(2)))  unsigned int  u32x2;
typedef __attribute__((ext_vector_type(4)))  float         f32x4;

#define NROWS   784
#define KDIM    1536
#define NMEM    100000
#define NTILE   64
#define NWG     224
#define NCHUNK  ((NMEM + NTILE - 1) / NTILE)   // 1563
#define KC      (KDIM / 32)                    // 48
#define MT      (NROWS / 16)                   // 49
#define BSTRIDE 1544                           // 1536 + 8 halves pad (bank-conflict-free)
#define INF_F   3.0e38f

struct B32B { u32x4 a, b; };

__device__ __forceinline__ unsigned short f2bf(float f) {
  unsigned int u = __builtin_bit_cast(unsigned int, f);
  unsigned int r = u + 0x7FFFu + ((u >> 16) & 1u);   // RNE truncate f32 -> bf16
  return (unsigned short)(r >> 16);
}

// ---------------------------------------------------------------- embedding
// avg_pool3(f2) concat upsample2x(avg_pool3(f3)) -> emb[784][1536] (row-major)
__global__ void embed_kernel(const float* __restrict__ f2,
                             const float* __restrict__ f3,
                             float* __restrict__ emb) {
  int idx = blockIdx.x * 256 + threadIdx.x;     // exactly 784*1536 threads
  int m  = idx / KDIM;
  int ch = idx - m * KDIM;
  int y = m / 28, x = m - y * 28;
  float s = 0.f;
  if (ch < 512) {
    const float* p = f2 + (size_t)ch * 784;
    for (int dy = -1; dy <= 1; ++dy) {
      int yy = y + dy; if (yy < 0 || yy > 27) continue;
      for (int dx = -1; dx <= 1; ++dx) {
        int xx = x + dx; if (xx < 0 || xx > 27) continue;
        s += p[yy * 28 + xx];
      }
    }
  } else {
    int cc = ch - 512;
    int Y = y >> 1, X = x >> 1;
    const float* p = f3 + (size_t)cc * 196;
    for (int dy = -1; dy <= 1; ++dy) {
      int yy = Y + dy; if (yy < 0 || yy > 13) continue;
      for (int dx = -1; dx <= 1; ++dx) {
        int xx = X + dx; if (xx < 0 || xx > 13) continue;
        s += p[yy * 14 + xx];
      }
    }
  }
  emb[idx] = s * (1.f / 9.f);
}

// ---------------------------------------------------------------- ||e||^2
__global__ void norme_kernel(const float* __restrict__ emb, float* __restrict__ normE) {
  __shared__ float red[256];
  int r = blockIdx.x;
  float s = 0.f;
  for (int k = threadIdx.x; k < KDIM; k += 256) {
    float v = emb[(size_t)r * KDIM + k];
    s += v * v;
  }
  red[threadIdx.x] = s; __syncthreads();
  for (int st = 128; st > 0; st >>= 1) {
    if (threadIdx.x < st) red[threadIdx.x] += red[threadIdx.x + st];
    __syncthreads();
  }
  if (threadIdx.x == 0) normE[r] = red[0];
}

// ---------------------------------------------------------------- A -> bf16 WMMA-fragment swizzle
// Layout [mt][kc][lane][j] : lane<16 -> M=lane, halves j = K {0..7,16..23}
//                            lane>=16 -> M=lane-16, halves j = K {8..15,24..31}
__global__ void aswz_kernel(const float* __restrict__ emb, unsigned short* __restrict__ A) {
  int idx = blockIdx.x * 256 + threadIdx.x;     // exactly 49*48*32*16 threads
  int j    = idx & 15;
  int lane = (idx >> 4) & 31;
  int t    = idx >> 9;                          // mt*48 + kc
  int kc   = t % KC;
  int mt   = t / KC;
  int row  = mt * 16 + (lane & 15);
  int k    = kc * 32 + (j & 7) + ((j >> 3) << 4) + ((lane >> 4) << 3);
  A[idx] = f2bf(emb[(size_t)row * KDIM + k]);
}

// ---------------------------------------------------------------- fused GEMM + top-9
__global__ __launch_bounds__(256) void gemm_topk_kernel(const float* __restrict__ mbank,
                                                        const unsigned short* __restrict__ Aswz,
                                                        float* __restrict__ partials) {
  extern __shared__ char smem[];
  unsigned short* Bs   = (unsigned short*)smem;                       // 64*1544 halves
  float* normM = (float*)(smem + (size_t)NTILE * BSTRIDE * 2);        // 64
  float* d2t   = normM + NTILE;                                       // 2 * 16*65
  float* top9  = d2t + 2 * 16 * 65;                                   // 784*9
  float* worst = top9 + NROWS * 9;                                    // 784

  int tid  = threadIdx.x;
  int wgid = blockIdx.x;

  for (int i = tid; i < NROWS * 9; i += 256) top9[i] = INF_F;
  for (int i = tid; i < NROWS;     i += 256) worst[i] = INF_F;

  int lane = tid & 31;
  int half = tid >> 7;                       // waves 0-3 -> mt 0..24, waves 4-7 -> mt 25..48
  int nloc = ((tid >> 5) & 3) * 16 + (lane & 15);
  int hi   = lane >> 4;

  for (int c = wgid; c < NCHUNK; c += NWG) {
    __syncthreads();
    int n0 = c * NTILE;
    if (tid < NTILE) normM[tid] = 0.f;
    __syncthreads();

    // ---- load 64 memory-bank rows (f32) -> bf16 LDS, norms via ds_add_f32
    const float* src = mbank + (size_t)n0 * KDIM;
    long remElems = (long)(NMEM - n0) * KDIM;
    for (int i = tid; i < (NTILE * KDIM) / 4; i += 256) {
      long e = (long)i * 4;
      f32x4 v; v.x = v.y = v.z = v.w = 0.f;
      if (e < remElems) v = *(const f32x4*)(src + e);
      int row = (int)(e / KDIM);
      int k   = (int)(e - (long)row * KDIM);
      atomicAdd(&normM[row], v.x * v.x + v.y * v.y + v.z * v.z + v.w * v.w);
      u32x2 st;
      st.x = (unsigned int)f2bf(v.x) | ((unsigned int)f2bf(v.y) << 16);
      st.y = (unsigned int)f2bf(v.z) | ((unsigned int)f2bf(v.w) << 16);
      *(u32x2*)(Bs + row * BSTRIDE + k) = st;
    }
    __syncthreads();
    if (tid < NTILE && (n0 + tid) >= NMEM) normM[tid] = INF_F;   // pad rows never selected
    __syncthreads();

    // ---- sweep M tiles: WMMA bf16 -> f32, emit -2*dot + ||m||^2, keep top-9
    for (int i = 0; i < 25; ++i) {
      int mt = half * 25 + i;
      bool act = (mt < MT);
      if (act) {
        v8f acc = {};
        const u32x4* aBase = ((const u32x4*)Aswz) + ((size_t)(mt * KC) * 32 + lane) * 2;
        const u32x4* bBase = ((const u32x4*)Bs) + (nloc * (BSTRIDE / 8) + hi);
        for (int kc = 0; kc < KC; ++kc) {
          B32B au; au.a = aBase[kc * 64]; au.b = aBase[kc * 64 + 1];
          B32B bu; bu.a = bBase[kc * 4];  bu.b = bBase[kc * 4 + 2];
          v16bf av = __builtin_bit_cast(v16bf, au);
          v16bf bv = __builtin_bit_cast(v16bf, bu);
          acc = __builtin_amdgcn_wmma_f32_16x16x32_bf16(false, av, false, bv,
                                                        (short)0, acc, false, false);
        }
        float nm = normM[nloc];
        int m8 = hi * 8;
        #pragma unroll
        for (int r = 0; r < 8; ++r)
          d2t[half * 1040 + (m8 + r) * 65 + nloc] = fmaf(-2.f, acc[r], nm);
      }
      __syncthreads();
      if (tid < 32) {
        int h2 = tid >> 4;
        int rr = tid & 15;
        int mt2 = h2 * 25 + i;
        if (mt2 < MT) {
          int row = mt2 * 16 + rr;
          float wst = worst[row];
          float* t9 = &top9[row * 9];
          for (int c2 = 0; c2 < NTILE; ++c2) {
            float v = d2t[h2 * 1040 + rr * 65 + c2];
            if (v < wst) {
              int mi = 0; float mv = t9[0];
              #pragma unroll
              for (int j = 1; j < 9; ++j) { float tv = t9[j]; if (tv > mv) { mv = tv; mi = j; } }
              t9[mi] = v;
              mv = t9[0];
              #pragma unroll
              for (int j = 1; j < 9; ++j) mv = fmaxf(mv, t9[j]);
              wst = mv;
            }
          }
          worst[row] = wst;
        }
      }
      __syncthreads();
    }
  }
  __syncthreads();
  float* dst = partials + (size_t)wgid * NROWS * 9;
  for (int i = tid; i < NROWS * 9; i += 256) dst[i] = top9[i];
}

// ---------------------------------------------------------------- merge partial top-9, sqrt, sort
__global__ void merge_topk_kernel(const float* __restrict__ partials,
                                  const float* __restrict__ normE,
                                  float* __restrict__ scores,
                                  float* __restrict__ amap) {
  int r = blockIdx.x * blockDim.x + threadIdx.x;
  if (r >= NROWS) return;
  float t9[9];
  #pragma unroll
  for (int j = 0; j < 9; ++j) t9[j] = INF_F;
  float wst = INF_F;
  for (int w = 0; w < NWG; ++w) {
    const float* p = partials + ((size_t)w * NROWS + r) * 9;
    for (int j = 0; j < 9; ++j) {
      float v = p[j];
      if (v < wst) {
        int mi = 0; float mv = t9[0];
        for (int jj = 1; jj < 9; ++jj) { if (t9[jj] > mv) { mv = t9[jj]; mi = jj; } }
        t9[mi] = v;
        mv = t9[0];
        for (int jj = 1; jj < 9; ++jj) mv = fmaxf(mv, t9[jj]);
        wst = mv;
      }
    }
  }
  float ne = normE[r];
  float s[9];
  for (int j = 0; j < 9; ++j) s[j] = sqrtf(fmaxf(t9[j] + ne, 1e-12f));
  for (int a = 1; a < 9; ++a) {            // ascending like top_k(-d2)
    float key = s[a]; int b = a - 1;
    while (b >= 0 && s[b] > key) { s[b + 1] = s[b]; --b; }
    s[b + 1] = key;
  }
  for (int j = 0; j < 9; ++j) scores[r * 9 + j] = s[j];
  amap[r] = s[0];
}

// ---------------------------------------------------------------- anomaly score
__global__ void score_kernel(const float* __restrict__ scores,
                             const float* __restrict__ amap,
                             float* __restrict__ outScore) {
  __shared__ float vmax[256]; __shared__ int vidx[256];
  int tid = threadIdx.x;
  float m = -INF_F; int mi = 0;
  for (int r = tid; r < NROWS; r += 256) {
    float v = amap[r];
    if (v > m) { m = v; mi = r; }
  }
  vmax[tid] = m; vidx[tid] = mi; __syncthreads();
  for (int st = 128; st > 0; st >>= 1) {
    if (tid < st && vmax[tid + st] > vmax[tid]) { vmax[tid] = vmax[tid + st]; vidx[tid] = vidx[tid + st]; }
    __syncthreads();
  }
  if (tid == 0) {
    const float* c = scores + (size_t)vidx[0] * 9;
    float emax = -INF_F, esum = 0.f;
    for (int j = 0; j < 9; ++j) { float e = expf(c[j]); esum += e; emax = fmaxf(emax, e); }
    outScore[0] = (1.f - emax / esum) * vmax[0];
  }
}

// ---------------------------------------------------------------- bilinear 28 -> 224 (half-pixel)
__global__ void bilinear_kernel(const float* __restrict__ amap, float* __restrict__ outv) {
  int idx = blockIdx.x * 256 + threadIdx.x;     // exactly 224*224 threads
  int y = idx / 224, x = idx - y * 224;
  float sy = (y + 0.5f) * 0.125f - 0.5f;
  float sx = (x + 0.5f) * 0.125f - 0.5f;
  float fy0 = floorf(sy), fx0 = floorf(sx);
  float wy = sy - fy0, wx = sx - fx0;
  int y0 = (int)fy0, x0 = (int)fx0;
  int y0c = min(max(y0, 0), 27),     y1c = min(max(y0 + 1, 0), 27);
  int x0c = min(max(x0, 0), 27),     x1c = min(max(x0 + 1, 0), 27);
  float a = amap[y0c * 28 + x0c], b = amap[y0c * 28 + x1c];
  float cc = amap[y1c * 28 + x0c], d = amap[y1c * 28 + x1c];
  float top = a + wx * (b - a), bot = cc + wx * (d - cc);
  outv[idx] = top + wy * (bot - top);
}

// ---------------------------------------------------------------- separable Gaussian (sigma=4, R=16, symmetric pad)
__global__ void gauss_kernel(const float* __restrict__ in, float* __restrict__ outv, int vertical) {
  __shared__ float kw[33];
  int tid = threadIdx.x;
  if (tid < 33) { float d = (float)(tid - 16); kw[tid] = expf(-d * d * (1.0f / 32.0f)); }
  __syncthreads();
  float norm = 0.f;
  #pragma unroll
  for (int i = 0; i < 33; ++i) norm += kw[i];
  float inv = 1.f / norm;
  int idx = blockIdx.x * 256 + tid;             // exactly 224*224 threads
  int y = idx / 224, x = idx - y * 224;
  float s = 0.f;
  for (int i = -16; i <= 16; ++i) {
    int p = (vertical ? y : x) + i;
    p = (p < 0) ? (-1 - p) : ((p > 223) ? (447 - p) : p);
    float v = vertical ? in[p * 224 + x] : in[y * 224 + p];
    s += kw[i + 16] * v;
  }
  outv[idx] = s * inv;
}

// ---------------------------------------------------------------- launcher
extern "C" void kernel_launch(void* const* d_in, const int* in_sizes, int n_in,
                              void* d_out, int out_size, void* d_ws, size_t ws_size,
                              hipStream_t stream) {
  (void)in_sizes; (void)n_in; (void)out_size; (void)ws_size;
  const float* f2 = (const float*)d_in[0];   // (512,28,28)
  const float* f3 = (const float*)d_in[1];   // (1024,14,14)
  const float* mb = (const float*)d_in[2];   // (100000,1536)
  float* out = (float*)d_out;                // 224*224 map + 1 score

  char* ws = (char*)d_ws;
  float* emb            = (float*)ws;           ws += (size_t)NROWS * KDIM * 4;
  unsigned short* Aswz  = (unsigned short*)ws;  ws += (size_t)NROWS * KDIM * 2;
  float* normE          = (float*)ws;           ws += (size_t)NROWS * 4;
  float* partials       = (float*)ws;           ws += (size_t)NWG * NROWS * 9 * 4;
  float* scores         = (float*)ws;           ws += (size_t)NROWS * 9 * 4;
  float* amap           = (float*)ws;           ws += (size_t)NROWS * 4;
  float* resized        = (float*)ws;           ws += (size_t)224 * 224 * 4;
  float* tmp            = (float*)ws;

  embed_kernel<<<(NROWS * KDIM) / 256, 256, 0, stream>>>(f2, f3, emb);
  norme_kernel<<<NROWS, 256, 0, stream>>>(emb, normE);
  aswz_kernel<<<(NROWS * KDIM) / 256, 256, 0, stream>>>(emb, Aswz);

  size_t smem = (size_t)NTILE * BSTRIDE * 2 +
                (NTILE + 2 * 16 * 65 + NROWS * 9 + NROWS) * sizeof(float);
  gemm_topk_kernel<<<NWG, 256, smem, stream>>>(mb, Aswz, partials);

  merge_topk_kernel<<<(NROWS + 127) / 128, 128, 0, stream>>>(partials, normE, scores, amap);
  score_kernel<<<1, 256, 0, stream>>>(scores, amap, out + 224 * 224);
  bilinear_kernel<<<(224 * 224) / 256, 256, 0, stream>>>(amap, resized);
  gauss_kernel<<<(224 * 224) / 256, 256, 0, stream>>>(resized, tmp, 0);
  gauss_kernel<<<(224 * 224) / 256, 256, 0, stream>>>(tmp, out, 1);
}
